// Attention_68779606278882
// MI455X (gfx1250) — matrix operand order repaired and checked by reading, not compiled, not run on hardware
//
#include <hip/hip_runtime.h>
#include <hip/hip_bf16.h>

typedef __bf16 bf16;
typedef __attribute__((ext_vector_type(16))) __bf16 v16bf;
typedef __attribute__((ext_vector_type(8)))  __bf16 v8bf;
typedef __attribute__((ext_vector_type(8)))  float  v8f;
typedef __attribute__((ext_vector_type(4))) unsigned int u32x4;
typedef __attribute__((ext_vector_type(8))) int          i32x8;
typedef __attribute__((ext_vector_type(4))) int          i32x4;

#define DIMX   2048
#define HEADS  16
#define DHEAD  128
#define MEMLEN 256
#define NQ     2048
#define KVLEN  2304   // NQ + MEMLEN

// Tensor Data Mover path (gfx1250; this toolchain exposes the 6-arg builtin).
#if defined(__gfx1250__) && __has_builtin(__builtin_amdgcn_tensor_load_to_lds) && \
    __has_builtin(__builtin_amdgcn_s_wait_tensorcnt)
#define USE_TDM 1
#endif

// ---------------------------------------------------------------------------
// TDM descriptor builders (CDNA5 ISA §8.3 / §8.4). 2-byte elements, 2-D tile.
// group0: [1:0]=count=1, [63:32]=lds_addr, [120:64]=global_addr, [127:126]=type=2
// group1: [17:16]=data_size(1->2B), [79:48]=tensor_dim0, [111:80]=tensor_dim1,
//         [127:112]=tile_dim0, [143:128]=tile_dim1, [207:160]=tensor_dim0_stride
// ---------------------------------------------------------------------------
__device__ __forceinline__ u32x4 tdm_group0(unsigned lds_addr, const void* gp) {
  const unsigned long long ga = (unsigned long long)(size_t)gp;
  u32x4 g;
  g[0] = 1u;                                   // count=1, user mode, no gather
  g[1] = lds_addr;                             // LDS byte address
  g[2] = (unsigned)ga;                         // global_addr[31:0]
  g[3] = (unsigned)((ga >> 32) & 0x1FFFFFFu) | 0x80000000u; // addr[56:32] | type=2
  return g;
}

__device__ __forceinline__ i32x8 tdm_group1(unsigned td0, unsigned td1,
                                            unsigned tile0, unsigned tile1,
                                            unsigned stride0) {
  i32x8 g;
  g[0] = 0x00010000;                           // workgroup_mask=0, data_size=1 (2B)
  g[1] = (int)((td0 & 0xFFFFu) << 16);         // tensor_dim0[15:0]
  g[2] = (int)((td0 >> 16) | ((td1 & 0xFFFFu) << 16));
  g[3] = (int)((td1 >> 16) | (tile0 << 16));   // tile_dim0
  g[4] = (int)(tile1 & 0xFFFFu);               // tile_dim1 (tile_dim2=0)
  g[5] = (int)stride0;                         // tensor_dim0_stride[31:0]
  g[6] = 0;                                    // stride0 hi / stride1 lo
  g[7] = 0;
  return g;
}

// ---------------------------------------------------------------------------
// RMSNorm: xn = x * rsqrt(sum(x^2)+eps) * sqrt(dim) * gamma  -> bf16
// ---------------------------------------------------------------------------
__global__ __launch_bounds__(256) void rmsnorm_kernel(const float* __restrict__ x,
                                                      const float* __restrict__ gamma,
                                                      bf16* __restrict__ xn) {
  __shared__ float red[256];
  const int row = blockIdx.x;
  const int tid = threadIdx.x;
  const float* xr = x + (size_t)row * DIMX;
  float s = 0.f;
  for (int c = tid; c < DIMX; c += 256) { float v = xr[c]; s += v * v; }
  red[tid] = s;
  __syncthreads();
  for (int off = 128; off > 0; off >>= 1) {
    if (tid < off) red[tid] += red[tid + off];
    __syncthreads();
  }
  const float scale = rsqrtf(red[0] + 1e-24f) * 45.254833995939045f; // sqrt(2048)
  for (int c = tid; c < DIMX; c += 256)
    xn[(size_t)row * DIMX + c] = (bf16)(xr[c] * scale * gamma[c]);
}

// ---------------------------------------------------------------------------
// Weight transpose + cast: out[N][K] (bf16) = in[K][N] (f32). One-time pass so
// every GEMM B-tile becomes a contiguous rectangle (TDM/WMMA-fragment friendly).
// ---------------------------------------------------------------------------
__global__ __launch_bounds__(256) void transpose_cast_kernel(const float* __restrict__ in,
                                                             bf16* __restrict__ out,
                                                             int K, int N) {
  __shared__ bf16 tile[32][33];
  const int kb = blockIdx.y * 32;
  const int nb = blockIdx.x * 32;
  const int tx = threadIdx.x & 31;
  const int ty = threadIdx.x >> 5;    // 0..7
  for (int r = ty; r < 32; r += 8)
    tile[r][tx] = (bf16)in[(size_t)(kb + r) * N + nb + tx];
  __syncthreads();
  for (int r = ty; r < 32; r += 8)
    out[(size_t)(nb + r) * K + kb + tx] = tile[tx][r];
}

// ---------------------------------------------------------------------------
// bf16 GEMM: C(f32)[M,N] = A[M,K] * Bt[N,K]^T, WMMA f32_16x16x32_bf16.
// Block tile 128x128, K-step 32, 8 waves (2x4), each wave 64x32.
// Tiles staged into LDS by the Tensor Data Mover (wave 0), TENSORcnt-synced.
// ---------------------------------------------------------------------------
#define GM 128
#define GN 128
#define GK 32

__global__ __launch_bounds__(256) void gemm_bf16_kernel(const bf16* __restrict__ A,
                                                        const bf16* __restrict__ Bt,
                                                        float* __restrict__ C,
                                                        int M, int N, int K) {
  __shared__ bf16 As[GM][GK];        // 8 KB
  __shared__ bf16 Bs[GN][GK];        // 8 KB, Bs[n][k]
  const int tid  = threadIdx.x;
  const int lane = tid & 31;
  const int wave = tid >> 5;     // 0..7
  const int half = lane >> 4;    // hi/lo 16-lane group
  const int l16  = lane & 15;
  const int wm   = wave >> 2;    // 0..1 : 64-row slice
  const int wn   = wave & 3;     // 0..3 : 32-col slice
  const int row0 = blockIdx.y * GM;
  const int col0 = blockIdx.x * GN;

  v8f acc[4][2] = {};

#ifdef USE_TDM
  const unsigned lds_a = (unsigned)(size_t)&As[0][0];
  const unsigned lds_b = (unsigned)(size_t)&Bs[0][0];
  const i32x4 gz4 = {0, 0, 0, 0};
  const i32x8 gz8 = {0, 0, 0, 0, 0, 0, 0, 0};
  const i32x8 g1t = tdm_group1((unsigned)K, 0x40000000u, (unsigned)GK, (unsigned)GM,
                               (unsigned)K);
#endif

  for (int k0 = 0; k0 < K; k0 += GK) {
    __syncthreads();
#ifdef USE_TDM
    if (wave == 0) {
      __builtin_amdgcn_tensor_load_to_lds(
          tdm_group0(lds_a, A + (size_t)row0 * K + k0), g1t, gz4, gz4, gz8, 0);
      __builtin_amdgcn_tensor_load_to_lds(
          tdm_group0(lds_b, Bt + (size_t)col0 * K + k0), g1t, gz4, gz4, gz8, 0);
      __builtin_amdgcn_s_wait_tensorcnt((short)0);
    }
#else
    for (int t = tid; t < (GM * GK) / 8; t += 256) {
      int r = t >> 2;
      int c = (t & 3) * 8;
      *(uint4*)&As[r][c] = *(const uint4*)&A[(size_t)(row0 + r) * K + k0 + c];
      if (k0 + GK < K)
        __builtin_prefetch(&A[(size_t)(row0 + r) * K + k0 + GK + c], 0, 0);
    }
    for (int t = tid; t < (GN * GK) / 8; t += 256) {
      int r = t >> 2;
      int c = (t & 3) * 8;
      *(uint4*)&Bs[r][c] = *(const uint4*)&Bt[(size_t)(col0 + r) * K + k0 + c];
      if (k0 + GK < K)
        __builtin_prefetch(&Bt[(size_t)(col0 + r) * K + k0 + GK + c], 0, 0);
    }
#endif
    __syncthreads();

#pragma unroll
    for (int nt = 0; nt < 2; ++nt) {
      const int ncol = wn * 32 + nt * 16;
      // B frag: lane l16 = N column, element e -> K = half*16 + e (contiguous)
      v16bf bfrag = *(const v16bf*)&Bs[ncol + l16][half * 16];
#pragma unroll
      for (int mt = 0; mt < 4; ++mt) {
        const int mrow = wm * 64 + mt * 16 + l16;
        // A frag: lane l16 = row M, element e -> K = half*16 + e (contiguous)
        const bf16* p = &As[mrow][half * 16];
        v8bf lo = *(const v8bf*)p;
        v8bf hi = *(const v8bf*)(p + 8);
        v16bf afrag = __builtin_shufflevector(lo, hi,
            0, 1, 2, 3, 4, 5, 6, 7, 8, 9, 10, 11, 12, 13, 14, 15);
        acc[mt][nt] = __builtin_amdgcn_wmma_f32_16x16x32_bf16(
            false, afrag, false, bfrag, (short)0, acc[mt][nt], false, false);
      }
    }
  }
  // C frag: VGPR r -> row = r + half*8, col = l16
#pragma unroll
  for (int mt = 0; mt < 4; ++mt)
#pragma unroll
    for (int nt = 0; nt < 2; ++nt)
#pragma unroll
      for (int r = 0; r < 8; ++r) {
        int i = row0 + wm * 64 + mt * 16 + r + half * 8;
        int j = col0 + wn * 32 + nt * 16 + l16;
        C[(size_t)i * N + j] = acc[mt][nt][r];
      }
}

// ---------------------------------------------------------------------------
// Pack Q: rotary + fold 1/sqrt(d) scale, cast bf16, layout [h][n][d]
// ---------------------------------------------------------------------------
__global__ __launch_bounds__(256) void pack_q_kernel(const float* __restrict__ qf,
                                                     const float* __restrict__ q_rot,
                                                     bf16* __restrict__ Qb) {
  int idx = blockIdx.x * 256 + threadIdx.x;       // hh * 2^18 + i * 128 + d
  if (idx >= HEADS * NQ * DHEAD) return;
  const int d  = idx & 127;
  const int i  = (idx >> 7) & (NQ - 1);
  const int hh = idx >> 18;
  const float t  = qf[(size_t)i * DIMX + hh * DHEAD + d];
  const int   dp = (d < 64) ? d + 64 : d - 64;
  const float tp = qf[(size_t)i * DIMX + hh * DHEAD + dp];
  const float rh = (d < 64) ? -tp : tp;
  const float pos = q_rot[i * DHEAD + d];
  const float val = (t * __cosf(pos) + rh * __sinf(pos)) * 0.08838834764831845f;
  Qb[idx] = (bf16)val;
}

// ---------------------------------------------------------------------------
// Pack K/V: XL-cache concat + rotary(K) -> K bf16 [h][kv][d]; V stored
// head-transposed bf16 [h][d][kv] so flash V-tiles are contiguous rectangles.
// Also writes next_xl_memories (pre-rotary, [2][16][2048][128]) into d_out.
// ---------------------------------------------------------------------------
__global__ __launch_bounds__(256) void pack_kv_kernel(const float* __restrict__ kvf,
                                                      const float* __restrict__ xl,
                                                      const float* __restrict__ k_rot,
                                                      bf16* __restrict__ Kb,
                                                      bf16* __restrict__ Vb,
                                                      float* __restrict__ next_xl) {
  int idx = blockIdx.x * 256 + threadIdx.x;
  if (idx >= HEADS * KVLEN * DHEAD) return;
  const int d   = idx & 127;
  const int rem = idx % (KVLEN * DHEAD);
  const int j   = rem >> 7;
  const int hh  = idx / (KVLEN * DHEAD);
  const int dp  = (d < 64) ? d + 64 : d - 64;
  float kval, kpart, vval;
  if (j < MEMLEN) {
    const float* kx = xl + ((size_t)hh * MEMLEN + j) * DHEAD;
    kval  = kx[d];
    kpart = kx[dp];
    vval  = xl[((size_t)(HEADS + hh) * MEMLEN + j) * DHEAD + d];
  } else {
    const int i = j - MEMLEN;
    kval  = kvf[(size_t)i * (2 * DIMX) + hh * DHEAD + d];
    kpart = kvf[(size_t)i * (2 * DIMX) + hh * DHEAD + dp];
    vval  = kvf[(size_t)i * (2 * DIMX) + DIMX + hh * DHEAD + d];
    next_xl[((size_t)hh * NQ + i) * DHEAD + d] = kval;
    next_xl[(size_t)HEADS * NQ * DHEAD + ((size_t)hh * NQ + i) * DHEAD + d] = vval;
  }
  const float rh  = (d < 64) ? -kpart : kpart;
  const float pos = k_rot[j * DHEAD + d];
  Kb[idx] = (bf16)(kval * __cosf(pos) + rh * __sinf(pos));
  Vb[((size_t)hh * DHEAD + d) * KVLEN + j] = (bf16)vval;   // [h][d][j]
}

// ---------------------------------------------------------------------------
// Flash attention (causal bottom-right, offset = KVLEN-NQ = 256).
// Block = 8 waves x 16 q-rows = 128 rows, one head per blockIdx.y.
// S = Q*K^T via WMMA (K-frags contiguous from global), streaming softmax via
// LDS, O += P*V via WMMA; V tile ([d][j] rectangle) staged into LDS by TDM.
// ---------------------------------------------------------------------------
#define FA_BM 128
#define FA_BJ 64

__global__ __launch_bounds__(256) void flash_attn_kernel(const bf16* __restrict__ Q,
                                                         const bf16* __restrict__ K,
                                                         const bf16* __restrict__ V,
                                                         bf16* __restrict__ Obf) {
  __shared__ float Sbuf[8][16][FA_BJ];   // 32 KB; P(bf16) aliased at row base
  __shared__ bf16  Vt[DHEAD][FA_BJ];     // 16 KB, Vt[d][j]
  __shared__ float mrow[8][16], lrow[8][16], arow[8][16];

  const int tid  = threadIdx.x;
  const int lane = tid & 31;
  const int wave = tid >> 5;
  const int half = lane >> 4;
  const int l16  = lane & 15;
  const int hh   = blockIdx.y;
  const int i0   = blockIdx.x * FA_BM;
  const int iw   = i0 + wave * 16;       // this wave's first q row

  const bf16* Qh = Q + (size_t)hh * NQ * DHEAD;
  const bf16* Kh = K + (size_t)hh * KVLEN * DHEAD;
  const bf16* Vh = V + (size_t)hh * DHEAD * KVLEN;   // [d][j]

  // Q fragments for this wave's 16 rows (4 d-steps of 32)
  v16bf qa[4];
#pragma unroll
  for (int dt = 0; dt < 4; ++dt) {
    const bf16* p = Qh + (size_t)(iw + l16) * DHEAD + dt * 32 + half * 16;
    v8bf lo = *(const v8bf*)p;
    v8bf hi = *(const v8bf*)(p + 8);
    qa[dt] = __builtin_shufflevector(lo, hi,
        0, 1, 2, 3, 4, 5, 6, 7, 8, 9, 10, 11, 12, 13, 14, 15);
  }

  v8f o[8] = {};
  if (lane < 16) {
    mrow[wave][lane] = -__builtin_inff();
    lrow[wave][lane] = 0.f;
    arow[wave][lane] = 1.f;
  }

#ifdef USE_TDM
  const unsigned lds_v = (unsigned)(size_t)&Vt[0][0];
  const i32x4 gz4 = {0, 0, 0, 0};
  const i32x8 gz8 = {0, 0, 0, 0, 0, 0, 0, 0};
  const i32x8 g1v = tdm_group1((unsigned)KVLEN, 0x40000000u, (unsigned)FA_BJ,
                               (unsigned)DHEAD, (unsigned)KVLEN);
#endif

  const int jend = i0 + FA_BM - 1 + (KVLEN - NQ) + 1;   // multiple of 64, <= KVLEN

  for (int j0 = 0; j0 < jend; j0 += FA_BJ) {
    __syncthreads();
    // Stage V tile Vt[d][j0..j0+63]
#ifdef USE_TDM
    if (wave == 0) {
      __builtin_amdgcn_tensor_load_to_lds(tdm_group0(lds_v, Vh + j0), g1v,
                                          gz4, gz4, gz8, 0);
      __builtin_amdgcn_s_wait_tensorcnt((short)0);
    }
#else
    for (int t = tid; t < (DHEAD * FA_BJ) / 8; t += 256) {
      int dd = t >> 3;
      int cc = (t & 7) * 8;
      *(uint4*)&Vt[dd][cc] = *(const uint4*)&Vh[(size_t)dd * KVLEN + j0 + cc];
    }
#endif
    // S = Q * K^T for 4 column tiles (K-frag contiguous over d in global)
    v8f s[4];
#pragma unroll
    for (int nt = 0; nt < 4; ++nt) {
      v8f c = {};
      const int j = j0 + nt * 16 + l16;
#pragma unroll
      for (int dt = 0; dt < 4; ++dt) {
        v16bf bfrag = *(const v16bf*)(Kh + (size_t)j * DHEAD + dt * 32 + half * 16);
        c = __builtin_amdgcn_wmma_f32_16x16x32_bf16(
            false, qa[dt], false, bfrag, (short)0, c, false, false);
      }
      s[nt] = c;
    }
    // Spill S to LDS (C-frag layout: row = r + half*8, col = l16)
#pragma unroll
    for (int nt = 0; nt < 4; ++nt)
#pragma unroll
      for (int r = 0; r < 8; ++r)
        Sbuf[wave][r + half * 8][nt * 16 + l16] = s[nt][r];
    __syncthreads();

    // Online softmax: lanes 0..15 each own one row
    if (lane < 16) {
      const int row  = lane;
      const int jlim = iw + row + (KVLEN - NQ);   // causal: j <= jlim
      const float m_old = mrow[wave][row];
      float mx = m_old;
      for (int c = 0; c < FA_BJ; ++c) {
        float sv = (j0 + c <= jlim) ? Sbuf[wave][row][c] : -__builtin_inff();
        mx = fmaxf(mx, sv);
      }
      float sum = 0.f;
      bf16* prow = (bf16*)&Sbuf[wave][row][0];   // P overwrites S row (safe: 2c < 4c)
      for (int c = 0; c < FA_BJ; ++c) {
        float sv = Sbuf[wave][row][c];
        float p = (j0 + c <= jlim && mx > -__builtin_inff()) ? __expf(sv - mx) : 0.f;
        prow[c] = (bf16)p;
        sum += p;
      }
      const float alpha = (m_old > -__builtin_inff()) ? __expf(m_old - mx) : 1.f;
      lrow[wave][row] = lrow[wave][row] * alpha + sum;
      mrow[wave][row] = mx;
      arow[wave][row] = alpha;
    }
    __syncthreads();

    // Rescale accumulators
#pragma unroll
    for (int dt = 0; dt < 8; ++dt)
#pragma unroll
      for (int r = 0; r < 8; ++r)
        o[dt][r] *= arow[wave][r + half * 8];

    // O += P * V  (contract over j: 2 k-steps of 32; 8 d tiles)
#pragma unroll
    for (int kk = 0; kk < 2; ++kk) {
      const bf16* prow = (const bf16*)&Sbuf[wave][l16][0];
      v8bf lo = *(const v8bf*)(prow + kk * 32 + half * 16);
      v8bf hi = *(const v8bf*)(prow + kk * 32 + half * 16 + 8);
      v16bf pa = __builtin_shufflevector(lo, hi,
          0, 1, 2, 3, 4, 5, 6, 7, 8, 9, 10, 11, 12, 13, 14, 15);
#pragma unroll
      for (int dt = 0; dt < 8; ++dt) {
        v16bf bfrag = *(const v16bf*)&Vt[dt * 16 + l16][kk * 32 + half * 16];
        o[dt] = __builtin_amdgcn_wmma_f32_16x16x32_bf16(
            false, pa, false, bfrag, (short)0, o[dt], false, false);
      }
    }
  }
  __syncthreads();
  // Normalize and store bf16 [n][h*d] for the output projection
#pragma unroll
  for (int dt = 0; dt < 8; ++dt)
#pragma unroll
    for (int r = 0; r < 8; ++r) {
      const int row = r + half * 8;
      const float val = o[dt][r] / lrow[wave][row];
      const int i = iw + row;
      Obf[(size_t)i * DIMX + hh * DHEAD + dt * 16 + l16] = (bf16)val;
    }
}

// ---------------------------------------------------------------------------
// Launch
// ---------------------------------------------------------------------------
extern "C" void kernel_launch(void* const* d_in, const int* in_sizes, int n_in,
                              void* d_out, int out_size, void* d_ws, size_t ws_size,
                              hipStream_t stream) {
  const float* x      = (const float*)d_in[0];   // [1,2048,2048]
  const float* q_rot  = (const float*)d_in[1];   // [2048,128]
  const float* k_rot  = (const float*)d_in[2];   // [2304,128]
  /* mask d_in[3] is all-true -> unused */
  const float* xl     = (const float*)d_in[4];   // [2,1,16,256,128]
  const float* gamma  = (const float*)d_in[5];   // [2048]
  const float* Wq     = (const float*)d_in[6];   // [2048,2048]
  const float* Wkv    = (const float*)d_in[7];   // [2048,4096]
  const float* Wout   = (const float*)d_in[8];   // [2048,2048]

  float* out      = (float*)d_out;               // [2048,2048]
  float* next_xl  = out + (size_t)NQ * DIMX;     // [2,16,2048,128]

  char* ws = (char*)d_ws;
  bf16*  xn    = (bf16*) (ws);                          //  8 MiB: [2048][2048]
  bf16*  wqt   = (bf16*) (ws + ((size_t)8  << 20));     //  8 MiB: Wq^T  [2048][2048]
  bf16*  wkvt  = (bf16*) (ws + ((size_t)16 << 20));     // 16 MiB: Wkv^T [4096][2048]
  bf16*  woutt = (bf16*) (ws + ((size_t)32 << 20));     //  8 MiB: Wout^T[2048][2048]
  float* qf    = (float*)(ws + ((size_t)40 << 20));     // 16 MiB
  float* kvf   = (float*)(ws + ((size_t)56 << 20));     // 32 MiB
  bf16*  Qb    = (bf16*) (ws + ((size_t)88 << 20));     //  8 MiB: [h][n][d]
  bf16*  Kb    = (bf16*) (ws + ((size_t)96 << 20));     //  9 MiB: [h][kv][d]
  bf16*  Vb    = (bf16*) (ws + ((size_t)106 << 20));    //  9 MiB: [h][d][kv]
  bf16*  attnb = (bf16*) (ws + ((size_t)116 << 20));    //  8 MiB

  // 1. RMSNorm -> bf16 activations
  rmsnorm_kernel<<<NQ, 256, 0, stream>>>(x, gamma, xn);
  // 2. Weights: transpose + cast to bf16 (one-time, makes B-tiles rectangular)
  transpose_cast_kernel<<<dim3(DIMX / 32, DIMX / 32), 256, 0, stream>>>(Wq, wqt, DIMX, DIMX);
  transpose_cast_kernel<<<dim3(2 * DIMX / 32, DIMX / 32), 256, 0, stream>>>(Wkv, wkvt, DIMX, 2 * DIMX);
  transpose_cast_kernel<<<dim3(DIMX / 32, DIMX / 32), 256, 0, stream>>>(Wout, woutt, DIMX, DIMX);
  // 3. Q and KV projections (TDM-staged WMMA)
  gemm_bf16_kernel<<<dim3(DIMX / GN, NQ / GM), 256, 0, stream>>>(xn, wqt, qf, NQ, DIMX, DIMX);
  gemm_bf16_kernel<<<dim3(2 * DIMX / GN, NQ / GM), 256, 0, stream>>>(xn, wkvt, kvf, NQ, 2 * DIMX, DIMX);
  // 4. Rotary / XL-concat / next_xl_memories / bf16 packing
  pack_q_kernel<<<(HEADS * NQ * DHEAD) / 256, 256, 0, stream>>>(qf, q_rot, Qb);
  pack_kv_kernel<<<(HEADS * KVLEN * DHEAD) / 256, 256, 0, stream>>>(kvf, xl, k_rot, Kb, Vb, next_xl);
  // 5. Flash attention (WMMA both GEMMs, TDM-staged V tiles)
  flash_attn_kernel<<<dim3(NQ / FA_BM, HEADS), 256, 0, stream>>>(Qb, Kb, Vb, attnb);
  // 6. Output projection (WMMA) -> d_out
  gemm_bf16_kernel<<<dim3(DIMX / GN, NQ / GM), 256, 0, stream>>>(attnb, woutt, out, NQ, DIMX, DIMX);
}
